// TransformerEncoderLayer_78769700209300
// MI455X (gfx1250) — compile-verified, matrix operand order checked by reading, not run on hardware
//
#include <hip/hip_runtime.h>
#include <hip/hip_bf16.h>
#include <math.h>

// ---------------------------------------------------------------------------
// Problem constants (T=2048, B=2, E=1024, H=16, FF=4096, K=64)
// ---------------------------------------------------------------------------
#define TT 2048
#define BB 2
#define EE 1024
#define HH 16
#define FFD 4096
#define KD 64
#define MM (TT * BB)   // 4096 rows

typedef __attribute__((ext_vector_type(16))) __bf16 bf16x16;
typedef __attribute__((ext_vector_type(8)))  float  f32x8;

union Frag {
  bf16x16 v;
  unsigned short u[16];
  uint4 q2[2];
};

__device__ __forceinline__ unsigned short f2bf(float f) {
  unsigned int x = __float_as_uint(f);
  x += 0x7FFFu + ((x >> 16) & 1u);           // round-to-nearest-even
  return (unsigned short)(x >> 16);
}

// Load a 16x32 row-major bf16 tile as a WMMA bf16 A/B operand fragment.
// Lane l holds row (l&15); K halves selected by (l>>4):
//   element j (j<8)  -> K = koff + j
//   element j (j>=8) -> K = koff + 16 + (j-8)
// Two 16-byte loads per lane; no per-element work.
__device__ __forceinline__ Frag load_frag_bf16(const unsigned short* p, int ld, int lane) {
  const int r    = lane & 15;
  const int koff = (lane >> 4) << 3;
  const unsigned short* src = p + (long)r * ld + koff;
  Frag f;
  f.q2[0] = *(const uint4*)(src);
  f.q2[1] = *(const uint4*)(src + 16);
  return f;
}

__device__ __forceinline__ f32x8 wmma_bf16(const Frag& a, const Frag& b, f32x8 c) {
  return __builtin_amdgcn_wmma_f32_16x16x32_bf16(false, a.v, false, b.v, (short)0, c,
                                                 false, false);
}

// 16-lane (half-wave) reductions; rows of a C-fragment live across 16 lanes.
__device__ __forceinline__ float redmax16(float v) {
#pragma unroll
  for (int m = 1; m < 16; m <<= 1) v = fmaxf(v, __shfl_xor(v, m, 32));
  return v;
}
__device__ __forceinline__ float redsum16(float v) {
#pragma unroll
  for (int m = 1; m < 16; m <<= 1) v += __shfl_xor(v, m, 32);
  return v;
}

// ---------------------------------------------------------------------------
// bf16 NT GEMM:  C[M,N] = act( (A[M,Kd] * B[N,Kd]^T + bias) * outscale )
// A, B bf16; accumulate fp32. Block: 256 threads (8 waves);
// wave -> 32x32 patch; block -> 64x128.
// ---------------------------------------------------------------------------
template <bool RELU, bool OUT_BF16>
__global__ __launch_bounds__(256) void gemm_nt_bf16_kernel(
    const unsigned short* __restrict__ A, long lda,
    const unsigned short* __restrict__ Bw, long ldb,
    const float* __restrict__ bias, float outscale,
    void* __restrict__ Cout, long ldc, int Kd)
{
  const int lane = threadIdx.x & 31;
  const int wave = threadIdx.x >> 5;
  const int wr = wave >> 2;   // 0..1
  const int wc = wave & 3;    // 0..3
  const long row0 = (long)blockIdx.x * 64 + (long)wr * 32;
  const long col0 = (long)blockIdx.y * 128 + (long)wc * 32;

  f32x8 acc00 = {}, acc01 = {}, acc10 = {}, acc11 = {};

  for (int k0 = 0; k0 < Kd; k0 += 32) {
    Frag a0 = load_frag_bf16(A + row0 * lda + k0, (int)lda, lane);
    Frag a1 = load_frag_bf16(A + (row0 + 16) * lda + k0, (int)lda, lane);
    Frag b0 = load_frag_bf16(Bw + col0 * ldb + k0, (int)ldb, lane);
    Frag b1 = load_frag_bf16(Bw + (col0 + 16) * ldb + k0, (int)ldb, lane);
    acc00 = wmma_bf16(a0, b0, acc00);
    acc01 = wmma_bf16(a0, b1, acc01);
    acc10 = wmma_bf16(a1, b0, acc10);
    acc11 = wmma_bf16(a1, b1, acc11);
  }

  // Epilogue. C layout: lanes 0-15: VGPR e -> row e; lanes 16-31: row e+8.
  const int rshift = (lane >> 4) << 3;
  const int cl     = lane & 15;
#pragma unroll
  for (int ti = 0; ti < 2; ++ti) {
#pragma unroll
    for (int tj = 0; tj < 2; ++tj) {
      const f32x8 acc = (ti == 0) ? (tj == 0 ? acc00 : acc01)
                                  : (tj == 0 ? acc10 : acc11);
      const long col = col0 + tj * 16 + cl;
      const float bv = bias ? bias[col] : 0.0f;
#pragma unroll
      for (int e = 0; e < 8; ++e) {
        const long row = row0 + ti * 16 + e + rshift;
        float vv = (acc[e] + bv) * outscale;
        if (RELU) vv = fmaxf(vv, 0.0f);
        if (OUT_BF16)
          ((unsigned short*)Cout)[row * ldc + col] = f2bf(vv);
        else
          ((float*)Cout)[row * ldc + col] = vv;
      }
    }
  }
}

// ---------------------------------------------------------------------------
// Fused flash-attention: one wave = 16 query rows of one (b,h).
// grid.x = B*H (32), grid.y = T/64 (32), block = 128 (4 waves).
// q,k bf16 in [T,B,H,K] (row stride B*E = 2048); vt bf16 in [B,H,K,T].
// q is pre-scaled by 1/sqrt(K) in its projection epilogue. ctx out bf16.
// ---------------------------------------------------------------------------
__global__ __launch_bounds__(128) void attention_kernel(
    const unsigned short* __restrict__ q, const unsigned short* __restrict__ k,
    const unsigned short* __restrict__ vt, unsigned short* __restrict__ ctx)
{
  __shared__ __align__(16) unsigned short plds[4][16][32];   // per-wave P tiles

  const int lane = threadIdx.x & 31;
  const int wave = threadIdx.x >> 5;
  const int bh = blockIdx.x;          // b*H + h
  const int b  = bh >> 4;
  const int h  = bh & 15;
  const long t0 = (long)blockIdx.y * 64 + (long)wave * 16;

  const long ld = (long)BB * EE;      // 2048: stride between successive t/s rows
  const unsigned short* qbase  = q  + (long)b * EE + (long)h * KD + t0 * ld;
  const unsigned short* kbase  = k  + (long)b * EE + (long)h * KD;
  const unsigned short* vtbase = vt + ((long)bh * KD) * TT;

  // Q fragments: 16 rows x 64 head-dims = two K-chunks of 32
  Frag aq0 = load_frag_bf16(qbase + 0,  (int)ld, lane);
  Frag aq1 = load_frag_bf16(qbase + 32, (int)ld, lane);

  f32x8 ot0 = {}, ot1 = {}, ot2 = {}, ot3 = {};
  float m_run[8], l_run[8];
#pragma unroll
  for (int e = 0; e < 8; ++e) { m_run[e] = -1e30f; l_run[e] = 0.0f; }

  const int rshift = (lane >> 4) << 3;
  const int cl     = lane & 15;

  for (int s = 0; s < TT; s += 32) {
    // ---- scores: S(16x32) = Qtile(16x64) @ K(s..s+31, :)^T -----------------
    const unsigned short* kp0 = kbase + (long)s * ld;
    const unsigned short* kp1 = kbase + (long)(s + 16) * ld;
    Frag kb00 = load_frag_bf16(kp0 + 0,  (int)ld, lane);
    Frag kb01 = load_frag_bf16(kp0 + 32, (int)ld, lane);
    Frag kb10 = load_frag_bf16(kp1 + 0,  (int)ld, lane);
    Frag kb11 = load_frag_bf16(kp1 + 32, (int)ld, lane);

    f32x8 S0 = {}, S1 = {};
    S0 = wmma_bf16(aq0, kb00, S0);
    S0 = wmma_bf16(aq1, kb01, S0);
    S1 = wmma_bf16(aq0, kb10, S1);
    S1 = wmma_bf16(aq1, kb11, S1);

    // ---- online softmax ----------------------------------------------------
#pragma unroll
    for (int e = 0; e < 8; ++e) {
      const float rm  = fmaxf(redmax16(S0[e]), redmax16(S1[e]));
      const float mnw = fmaxf(m_run[e], rm);
      const float alpha = __expf(m_run[e] - mnw);
      const float p0 = __expf(S0[e] - mnw);
      const float p1 = __expf(S1[e] - mnw);
      const float rsum = redsum16(p0 + p1);
      l_run[e] = l_run[e] * alpha + rsum;
      m_run[e] = mnw;
      ot0[e] *= alpha; ot1[e] *= alpha; ot2[e] *= alpha; ot3[e] *= alpha;
      const int row = e + rshift;
      plds[wave][row][cl]      = f2bf(p0);
      plds[wave][row][16 + cl] = f2bf(p1);
    }
    asm volatile("s_wait_dscnt 0" ::: "memory");

    // P tile (16x32) back in A-fragment order
    Frag pa = load_frag_bf16(&plds[wave][0][0], 32, lane);

    // ---- ctx += P(16x32) @ V(s..s+31, 0..63) -------------------------------
    Frag vb0 = load_frag_bf16(vtbase + 0  * TT + s, TT, lane);
    Frag vb1 = load_frag_bf16(vtbase + 16 * TT + s, TT, lane);
    Frag vb2 = load_frag_bf16(vtbase + 32 * TT + s, TT, lane);
    Frag vb3 = load_frag_bf16(vtbase + 48 * TT + s, TT, lane);
    ot0 = wmma_bf16(pa, vb0, ot0);
    ot1 = wmma_bf16(pa, vb1, ot1);
    ot2 = wmma_bf16(pa, vb2, ot2);
    ot3 = wmma_bf16(pa, vb3, ot3);
  }

  // ---- normalize + store ctx[t, b, h, :] (bf16) ----------------------------
  unsigned short* cbase = ctx + (long)b * EE + (long)h * KD;
#pragma unroll
  for (int e = 0; e < 8; ++e) {
    const long t = t0 + e + rshift;
    const float inv = 1.0f / l_run[e];
    unsigned short* crow = cbase + t * ld;
    crow[0  + cl] = f2bf(ot0[e] * inv);
    crow[16 + cl] = f2bf(ot1[e] * inv);
    crow[32 + cl] = f2bf(ot2[e] * inv);
    crow[48 + cl] = f2bf(ot3[e] * inv);
  }
}

// ---------------------------------------------------------------------------
// bf16 v [T,B,H,K] -> bf16 vt [B,H,K,T]
// ---------------------------------------------------------------------------
__global__ __launch_bounds__(256) void transpose_v_kernel(
    const unsigned short* __restrict__ v, unsigned short* __restrict__ vt) {
  const long idx = (long)blockIdx.x * 256 + threadIdx.x;   // over B*H*K*T
  const int s  = (int)(idx & (TT - 1));
  const long r = idx >> 11;
  const int kk = (int)(r & (KD - 1));
  const int bh = (int)(r >> 6);
  const int b = bh >> 4, h = bh & 15;
  vt[idx] = v[(long)s * (BB * EE) + (long)b * EE + (long)h * KD + kk];
}

// fp32 -> bf16, 4 elements/thread (n multiple of 1024)
__global__ __launch_bounds__(256) void cvt_bf16_kernel(
    const float* __restrict__ in, unsigned short* __restrict__ out, long n) {
  const long i = ((long)blockIdx.x * 256 + threadIdx.x) * 4;
  if (i >= n) return;
  const float4 x = *(const float4*)(in + i);
  union { unsigned short us[4]; uint2 u2; } o;
  o.us[0] = f2bf(x.x); o.us[1] = f2bf(x.y); o.us[2] = f2bf(x.z); o.us[3] = f2bf(x.w);
  *(uint2*)(out + i) = o.u2;
}

// key input: bf16(state * keep), keep from padding mask [B,T] bool bytes
__global__ __launch_bounds__(256) void cvt_bf16_masked_kernel(
    const float* __restrict__ in, const unsigned char* __restrict__ mask,
    unsigned short* __restrict__ out) {
  const long i = ((long)blockIdx.x * 256 + threadIdx.x) * 4;   // over M*E
  const long m = i >> 10;                                      // row = t*B + b
  const int t = (int)(m >> 1), b = (int)(m & 1);
  const float keep = mask[b * TT + t] ? 0.0f : 1.0f;
  const float4 x = *(const float4*)(in + i);
  union { unsigned short us[4]; uint2 u2; } o;
  o.us[0] = f2bf(x.x * keep); o.us[1] = f2bf(x.y * keep);
  o.us[2] = f2bf(x.z * keep); o.us[3] = f2bf(x.w * keep);
  *(uint2*)(out + i) = o.u2;
}

// out = LayerNorm(x + res) * g + be ; one block per row of E=1024.
// Optionally also writes a bf16 copy (operand for the next GEMM).
__global__ __launch_bounds__(256) void add_ln_kernel(
    const float* __restrict__ x, const float* __restrict__ res,
    const float* __restrict__ g, const float* __restrict__ be,
    float* __restrict__ out, unsigned short* __restrict__ out_bf)
{
  __shared__ float sh[16];
  const long row = blockIdx.x;
  const int tid = threadIdx.x;
  const int lane = tid & 31, wave = tid >> 5;
  const float* xr = x + row * EE;
  const float* rr = res + row * EE;

  float vals[4];
  float s = 0.0f, s2 = 0.0f;
#pragma unroll
  for (int i = 0; i < 4; ++i) {
    const int j = tid + i * 256;
    const float v = xr[j] + rr[j];
    vals[i] = v; s += v; s2 += v * v;
  }
#pragma unroll
  for (int m = 1; m < 32; m <<= 1) {
    s  += __shfl_xor(s, m, 32);
    s2 += __shfl_xor(s2, m, 32);
  }
  if (lane == 0) { sh[wave] = s; sh[8 + wave] = s2; }
  __syncthreads();
  float ts = 0.0f, ts2 = 0.0f;
#pragma unroll
  for (int w = 0; w < 8; ++w) { ts += sh[w]; ts2 += sh[8 + w]; }
  const float mu  = ts * (1.0f / EE);
  const float var = ts2 * (1.0f / EE) - mu * mu;
  const float inv = rsqrtf(var + 1e-5f);
#pragma unroll
  for (int i = 0; i < 4; ++i) {
    const int j = tid + i * 256;
    const float o = (vals[i] - mu) * inv * g[j] + be[j];
    out[row * EE + j] = o;
    if (out_bf) out_bf[row * EE + j] = f2bf(o);
  }
}

// ---------------------------------------------------------------------------
// Host launcher
// ---------------------------------------------------------------------------
extern "C" void kernel_launch(void* const* d_in, const int* in_sizes, int n_in,
                              void* d_out, int out_size, void* d_ws, size_t ws_size,
                              hipStream_t stream) {
  (void)in_sizes; (void)n_in; (void)out_size; (void)ws_size;

  const float* state = (const float*)d_in[0];
  const unsigned char* mask = (const unsigned char*)d_in[1];
  const float* Wq = (const float*)d_in[2];  const float* bq = (const float*)d_in[3];
  const float* Wk = (const float*)d_in[4];  const float* bk = (const float*)d_in[5];
  const float* Wv = (const float*)d_in[6];  const float* bv = (const float*)d_in[7];
  const float* Wo = (const float*)d_in[8];  const float* bo = (const float*)d_in[9];
  const float* ln1g = (const float*)d_in[10]; const float* ln1b = (const float*)d_in[11];
  const float* W1 = (const float*)d_in[12]; const float* b1 = (const float*)d_in[13];
  const float* W2 = (const float*)d_in[14]; const float* b2 = (const float*)d_in[15];
  const float* ln2g = (const float*)d_in[16]; const float* ln2b = (const float*)d_in[17];

  typedef unsigned short u16;
  char* ws = (char*)d_ws;
  const long MB = 1L << 20;
  // bf16 activation/weight pool (disjoint lifetimes; aliases noted)
  u16* state_bf  = (u16*)(ws + 0 * MB);    // 8 MB  [dead after V gemm]
  u16* statek_bf = (u16*)(ws + 8 * MB);    // 8 MB  [dead after K gemm]
  u16* wq_bf     = (u16*)(ws + 16 * MB);   // 2 MB
  u16* wk_bf     = (u16*)(ws + 18 * MB);   // 2 MB
  u16* wv_bf     = (u16*)(ws + 20 * MB);   // 2 MB
  u16* wo_bf     = (u16*)(ws + 22 * MB);   // 2 MB
  u16* q_bf      = (u16*)(ws + 24 * MB);   // 8 MB  [dead after attention]
  u16* k_bf      = (u16*)(ws + 32 * MB);   // 8 MB  [dead after attention]
  u16* v_bf      = (u16*)(ws + 40 * MB);   // 8 MB  [dead after transpose]
  u16* vt_bf     = (u16*)(ws + 48 * MB);   // 8 MB  [dead after attention]
  u16* ctx_bf    = (u16*)(ws + 56 * MB);   // 8 MB  [dead after O-proj]
  float* attn    = (float*)(ws + 64 * MB); // 16 MB [dead after LN1]
  float* state1  = (float*)(ws + 80 * MB); // 16 MB [residual, lives to LN2]
  // aliases:
  u16* w1_bf     = state_bf;               // converted after V gemm
  u16* w2_bf     = statek_bf;              // converted after K gemm
  u16* state1_bf = q_bf;                   // written at LN1
  u16* hidden    = k_bf;                   // 32 MB region (k,v,vt,ctx) at FF1
  float* ff2     = attn;                   // reused after LN1

  const long nME = (long)MM * EE;   // 4 M
  const long nE2 = (long)EE * EE;   // 1 M
  const long nEF = (long)EE * FFD;  // 4 M

  // ---- one-time bf16 conversions ------------------------------------------
  cvt_bf16_kernel<<<nME / 1024, 256, 0, stream>>>(state, state_bf, nME);
  cvt_bf16_masked_kernel<<<nME / 1024, 256, 0, stream>>>(state, mask, statek_bf);
  cvt_bf16_kernel<<<nE2 / 1024, 256, 0, stream>>>(Wq, wq_bf, nE2);
  cvt_bf16_kernel<<<nE2 / 1024, 256, 0, stream>>>(Wk, wk_bf, nE2);
  cvt_bf16_kernel<<<nE2 / 1024, 256, 0, stream>>>(Wv, wv_bf, nE2);
  cvt_bf16_kernel<<<nE2 / 1024, 256, 0, stream>>>(Wo, wo_bf, nE2);

  // ---- QKV projections (1/sqrt(K) folded into Q's epilogue) ---------------
  dim3 gP(MM / 64, EE / 128);
  gemm_nt_bf16_kernel<false, true><<<gP, 256, 0, stream>>>(
      state_bf, EE, wq_bf, EE, bq, 0.125f, q_bf, EE, EE);
  gemm_nt_bf16_kernel<false, true><<<gP, 256, 0, stream>>>(
      statek_bf, EE, wk_bf, EE, bk, 1.0f, k_bf, EE, EE);
  gemm_nt_bf16_kernel<false, true><<<gP, 256, 0, stream>>>(
      state_bf, EE, wv_bf, EE, bv, 1.0f, v_bf, EE, EE);

  // FFN weights converted now (their buffers just freed up)
  cvt_bf16_kernel<<<nEF / 1024, 256, 0, stream>>>(W1, w1_bf, nEF);
  cvt_bf16_kernel<<<nEF / 1024, 256, 0, stream>>>(W2, w2_bf, nEF);

  transpose_v_kernel<<<(MM * EE) / 256, 256, 0, stream>>>(v_bf, vt_bf);

  attention_kernel<<<dim3(BB * HH, TT / 64), 128, 0, stream>>>(q_bf, k_bf, vt_bf, ctx_bf);

  gemm_nt_bf16_kernel<false, false><<<gP, 256, 0, stream>>>(
      ctx_bf, EE, wo_bf, EE, bo, 1.0f, attn, EE, EE);

  add_ln_kernel<<<MM, 256, 0, stream>>>(attn, state, ln1g, ln1b, state1, state1_bf);

  // ---- FFN -----------------------------------------------------------------
  dim3 gF1(MM / 64, FFD / 128);
  gemm_nt_bf16_kernel<true, true><<<gF1, 256, 0, stream>>>(
      state1_bf, EE, w1_bf, EE, b1, 1.0f, hidden, FFD, EE);
  gemm_nt_bf16_kernel<false, false><<<gP, 256, 0, stream>>>(
      hidden, FFD, w2_bf, FFD, b2, 1.0f, ff2, EE, FFD);

  add_ln_kernel<<<MM, 256, 0, stream>>>(ff2, state1, ln2g, ln2b, (float*)d_out,
                                        nullptr);
}
